// MHA_75204877353136
// MI455X (gfx1250) — compile-verified, matrix operand order checked by reading, not run on hardware
//
#include <hip/hip_runtime.h>
#include <hip/hip_bf16.h>

typedef __bf16 bf16_t;
typedef __attribute__((ext_vector_type(16))) __bf16 v16bf;
typedef __attribute__((ext_vector_type(8)))  __bf16 v8bf;
typedef __attribute__((ext_vector_type(8)))  float  v8f;

constexpr int H_   = 8;
constexpr int N_   = 4096;
constexpr int FIN_ = 512;
constexpr int HD_  = 64;
constexpr int FOUT_= 512;

// ---------- helpers ----------

__device__ inline v8f zero_v8f() {
  v8f z;
#pragma unroll
  for (int i = 0; i < 8; ++i) z[i] = 0.0f;
  return z;
}

// Load one 16x16x32 bf16 A/B fragment from a row-major row pointer.
// Elements 0..7  = K(k0..k0+7)      (VGPR0..3)
// Elements 8..15 = K(k0+16..k0+23)  (VGPR4..7)
__device__ inline v16bf frag_ld(const bf16_t* p) {
  v16bf r;
  ((v8bf*)&r)[0] = ((const v8bf*)p)[0];
  ((v8bf*)&r)[1] = ((const v8bf*)(p + 16))[0];
  return r;
}

__device__ inline v8f wmma_bf16(v16bf a, v16bf b, v8f c) {
  return __builtin_amdgcn_wmma_f32_16x16x32_bf16(
      /*neg_a=*/false, a, /*neg_b=*/false, b,
      /*c_mod=*/(short)0, c, /*reuse_a=*/false, /*reuse_b=*/false);
}

// Async global->LDS copy, 16 bytes per lane. Tracked with ASYNCcnt.
// lds_addr: low 32 bits of the generic pointer == LDS byte address.
__device__ inline void async_ld_b128(unsigned lds_addr, const void* gaddr) {
  asm volatile("global_load_async_to_lds_b128 %0, %1, off"
               :: "v"(lds_addr), "v"(gaddr)
               : "memory");
}
__device__ inline void wait_async_le4() {
  asm volatile("s_wait_asynccnt 0x4" ::: "memory");
}
__device__ inline void wait_async_0() {
  asm volatile("s_wait_asynccnt 0x0" ::: "memory");
}

// Reduce across the 16 lanes of each wave32 half (rows live per-half).
__device__ inline float rmax16(float v) {
  v = fmaxf(v, __shfl_xor(v, 1, 32));
  v = fmaxf(v, __shfl_xor(v, 2, 32));
  v = fmaxf(v, __shfl_xor(v, 4, 32));
  v = fmaxf(v, __shfl_xor(v, 8, 32));
  return v;
}
__device__ inline float radd16(float v) {
  v += __shfl_xor(v, 1, 32);
  v += __shfl_xor(v, 2, 32);
  v += __shfl_xor(v, 4, 32);
  v += __shfl_xor(v, 8, 32);
  return v;
}

// ---------- kernel 1: fused QKV projection (fp32 -> bf16) ----------
// Grid (N/128, H), block 256. Output: Qb[h][n][d] (pre-scaled 1/8),
// Kb[h][n][d], VTb[h][d][n] (transposed for PV B-fragments).
__global__ __launch_bounds__(256) void qkv_kernel(
    const float* __restrict__ X,
    const float* __restrict__ WQ, const float* __restrict__ WK,
    const float* __restrict__ WV,
    bf16_t* __restrict__ Qb, bf16_t* __restrict__ Kb, bf16_t* __restrict__ VTb)
{
  __shared__ alignas(16) bf16_t Xs[128][72];   // 18 KB
  __shared__ alignas(16) bf16_t Ws[192][72];   // 27 KB  (rows 0-63 Q, 64-127 K, 128-191 V)

  const int tid  = threadIdx.x;
  const int lane = tid & 31;
  const int w    = tid >> 5;
  const int ln   = lane & 15;
  const int hi8  = (lane >> 4) * 8;
  const int h    = blockIdx.y;
  const int q0   = blockIdx.x * 128;

  v8f C[12];
#pragma unroll
  for (int t = 0; t < 12; ++t) C[t] = zero_v8f();

  for (int ks = 0; ks < FIN_ / 64; ++ks) {
    const int k0 = ks * 64;
    __syncthreads();
    { // stage X slice [128][64] fp32 -> bf16
      const int row = tid >> 1, c0 = (tid & 1) * 32;
      const float4* s4 =
          (const float4*)(X + ((size_t)h * N_ + q0 + row) * FIN_ + k0 + c0);
#pragma unroll
      for (int i = 0; i < 8; ++i) {
        float4 v = s4[i];
        bf16_t* d = &Xs[row][c0 + i * 4];
        d[0] = (bf16_t)v.x; d[1] = (bf16_t)v.y;
        d[2] = (bf16_t)v.z; d[3] = (bf16_t)v.w;
      }
    }
    if (tid < 192) { // stage fused W slice [192][64]
      const int r = tid;
      const float* src = (r < 64)  ? WQ + ((size_t)h * HD_ + r) * FIN_ + k0
                       : (r < 128) ? WK + ((size_t)h * HD_ + (r - 64)) * FIN_ + k0
                                   : WV + ((size_t)h * HD_ + (r - 128)) * FIN_ + k0;
      const float4* s4 = (const float4*)src;
#pragma unroll
      for (int i = 0; i < 16; ++i) {
        float4 v = s4[i];
        bf16_t* d = &Ws[r][i * 4];
        d[0] = (bf16_t)v.x; d[1] = (bf16_t)v.y;
        d[2] = (bf16_t)v.z; d[3] = (bf16_t)v.w;
      }
    }
    __syncthreads();

    const v16bf a0 = frag_ld(&Xs[w * 16 + ln][hi8]);
    const v16bf a1 = frag_ld(&Xs[w * 16 + ln][32 + hi8]);
#pragma unroll
    for (int ct = 0; ct < 12; ++ct) {
      v16bf b0 = frag_ld(&Ws[ct * 16 + ln][hi8]);
      v16bf b1 = frag_ld(&Ws[ct * 16 + ln][32 + hi8]);
      C[ct] = wmma_bf16(a0, b0, C[ct]);
      C[ct] = wmma_bf16(a1, b1, C[ct]);
    }
  }

  const int qb = q0 + w * 16;
#pragma unroll
  for (int ct = 0; ct < 12; ++ct) {
    const int col = ct * 16 + ln;
#pragma unroll
    for (int r = 0; r < 8; ++r) {
      const int qq = qb + r + hi8;
      const float v = C[ct][r];
      if (ct < 4)
        Qb[((size_t)h * N_ + qq) * HD_ + col] = (bf16_t)(v * 0.125f); // 1/sqrt(64)
      else if (ct < 8)
        Kb[((size_t)h * N_ + qq) * HD_ + (col - 64)] = (bf16_t)v;
      else
        VTb[((size_t)h * HD_ + (col - 128)) * N_ + qq] = (bf16_t)v;
    }
  }
}

// ---------- kernel 2: flash attention ----------
// Grid (N/128, H), block 256 (8 waves x 16 query rows). One pass over mask.
// K/V^T staging is double-buffered through GLOBAL_LOAD_ASYNC_TO_LDS_B128,
// so the HBM round-trip for the next key block overlaps this block's
// WMMA + softmax work. ASYNCcnt retires in order: after issuing the next
// block's 4 per-lane copies, waiting asynccnt<=4 guarantees this block's
// 4 copies have landed.
__global__ __launch_bounds__(256) void attn_kernel(
    const int* __restrict__ mask,
    const bf16_t* __restrict__ Qb, const bf16_t* __restrict__ Kb,
    const bf16_t* __restrict__ VTb, bf16_t* __restrict__ Hcat)
{
  __shared__ alignas(16) bf16_t Ks[2][64][72];     // 18 KB  key-major [key][d]
  __shared__ alignas(16) bf16_t VTs[2][64][72];    // 18 KB  d-major   [d][key]
  __shared__ alignas(16) bf16_t Pw[8][16][72];     // 18 KB  per-wave P tiles

  const int tid  = threadIdx.x;
  const int lane = tid & 31;
  const int w    = tid >> 5;
  const int ln   = lane & 15;
  const int hi8  = (lane >> 4) * 8;
  const int h    = blockIdx.y;
  const int q0   = blockIdx.x * 128;
  const int qb   = q0 + w * 16;

  // staging addresses for this thread (row of 64, 16-element chunk)
  const int srow = tid >> 2;          // 0..63
  const int sc0  = (tid & 3) * 16;    // 0,16,32,48
  const bf16_t* kgbase = Kb  + ((size_t)h * N_  + srow) * HD_ + sc0;
  const bf16_t* vgbase = VTb + ((size_t)h * HD_ + srow) * N_  + sc0;
  unsigned klds[2], vlds[2];
#pragma unroll
  for (int b = 0; b < 2; ++b) {
    klds[b] = (unsigned)(uintptr_t)&Ks[b][srow][sc0];
    vlds[b] = (unsigned)(uintptr_t)&VTs[b][srow][sc0];
  }

  // Q fragments resident in registers (32 bf16 per lane).
  const bf16_t* qrow = Qb + ((size_t)h * N_ + qb + ln) * HD_;
  const v16bf qa0 = frag_ld(qrow + hi8);
  const v16bf qa1 = frag_ld(qrow + 32 + hi8);

  float mrow[8], lrow[8];
  v8f O[4];
#pragma unroll
  for (int r = 0; r < 8; ++r) { mrow[r] = -3.0e38f; lrow[r] = 0.0f; }
#pragma unroll
  for (int t = 0; t < 4; ++t) O[t] = zero_v8f();

  // prologue: issue async staging for block 0 into buffer 0
  {
    const bf16_t* kg = kgbase;                 // key0 = 0
    const bf16_t* vg = vgbase;
    async_ld_b128(klds[0],      kg);
    async_ld_b128(klds[0] + 16, kg + 8);
    async_ld_b128(vlds[0],      vg);
    async_ld_b128(vlds[0] + 16, vg + 8);
  }

  constexpr int NB = N_ / 64;
  for (int kb = 0; kb < NB; ++kb) {
    const int cur = kb & 1;
    const int key0 = kb * 64;

    // issue next block's staging into the other buffer (safe: that buffer's
    // last readers finished before the trailing barrier of iteration kb-1)
    if (kb + 1 < NB) {
      const int nxt = cur ^ 1;
      const bf16_t* kg = kgbase + (size_t)(key0 + 64) * HD_;
      const bf16_t* vg = vgbase + (key0 + 64);
      async_ld_b128(klds[nxt],      kg);
      async_ld_b128(klds[nxt] + 16, kg + 8);
      async_ld_b128(vlds[nxt],      vg);
      async_ld_b128(vlds[nxt] + 16, vg + 8);
      wait_async_le4();   // current block's copies have retired
    } else {
      wait_async_0();
    }
    __syncthreads();      // everyone's copies for block kb are visible

    // S = Q K^T  (4 key-tiles x 2 WMMA over d)
    v8f S[4];
#pragma unroll
    for (int nt = 0; nt < 4; ++nt) {
      v8f c = zero_v8f();
      v16bf b0 = frag_ld(&Ks[cur][nt * 16 + ln][hi8]);
      v16bf b1 = frag_ld(&Ks[cur][nt * 16 + ln][32 + hi8]);
      c = wmma_bf16(qa0, b0, c);
      c = wmma_bf16(qa1, b1, c);
      S[nt] = c;
    }

    // mask: streaming non-temporal reads (512 MB, single use)
#pragma unroll
    for (int r = 0; r < 8; ++r) {
      const int* mp =
          mask + ((size_t)h * N_ + (qb + r + hi8)) * (size_t)N_ + key0 + ln;
#pragma unroll
      for (int nt = 0; nt < 4; ++nt) {
        int mv = __builtin_nontemporal_load(mp + nt * 16);
        if (mv == 0) S[nt][r] = -1.0e30f;
      }
    }

    // online softmax (rows reduced across 16-lane halves)
#pragma unroll
    for (int r = 0; r < 8; ++r) {
      float bm = fmaxf(fmaxf(S[0][r], S[1][r]), fmaxf(S[2][r], S[3][r]));
      bm = rmax16(bm);
      const float mn = fmaxf(mrow[r], bm);
      const float sc = __expf(mrow[r] - mn);
      mrow[r] = mn;
      float bs = 0.0f;
#pragma unroll
      for (int nt = 0; nt < 4; ++nt) {
        float p = __expf(S[nt][r] - mn);
        Pw[w][r + hi8][nt * 16 + ln] = (bf16_t)p;
        bs += p;
      }
      bs = radd16(bs);
      lrow[r] = lrow[r] * sc + bs;
#pragma unroll
      for (int nd = 0; nd < 4; ++nd) O[nd][r] = O[nd][r] * sc;
    }

    // O += P V  (re-fragment P from per-wave LDS; B from V^T)
    const v16bf pa0 = frag_ld(&Pw[w][ln][hi8]);
    const v16bf pa1 = frag_ld(&Pw[w][ln][32 + hi8]);
#pragma unroll
    for (int nd = 0; nd < 4; ++nd) {
      v16bf b0 = frag_ld(&VTs[cur][nd * 16 + ln][hi8]);
      v16bf b1 = frag_ld(&VTs[cur][nd * 16 + ln][32 + hi8]);
      O[nd] = wmma_bf16(pa0, b0, O[nd]);
      O[nd] = wmma_bf16(pa1, b1, O[nd]);
    }

    // trailing barrier: all waves done reading buffer `cur` before the next
    // iteration's issue overwrites it
    __syncthreads();
  }

  // epilogue: normalize and write concat-head layout [N][H*HD] bf16
#pragma unroll
  for (int nd = 0; nd < 4; ++nd)
#pragma unroll
    for (int r = 0; r < 8; ++r) {
      float val = O[nd][r] / lrow[r];
      Hcat[(size_t)(qb + r + hi8) * (H_ * HD_) + h * HD_ + nd * 16 + ln] =
          (bf16_t)val;
    }
}

// ---------- kernel 3: output projection ----------
// Grid (N/128, FOUT/64), block 256. out = Hcat(bf16) @ W_O^T, fp32 out.
__global__ __launch_bounds__(256) void proj_kernel(
    const float* __restrict__ WO, const bf16_t* __restrict__ Hcat,
    float* __restrict__ out)
{
  __shared__ alignas(16) bf16_t WOs[64][264];  // 33 KB: 64 out-cols x 256 k (half)

  const int tid  = threadIdx.x;
  const int lane = tid & 31;
  const int w    = tid >> 5;
  const int ln   = lane & 15;
  const int hi8  = (lane >> 4) * 8;
  const int q0   = blockIdx.x * 128;
  const int cb   = blockIdx.y * 64;

  v8f C[4];
#pragma unroll
  for (int t = 0; t < 4; ++t) C[t] = zero_v8f();

  const bf16_t* arow = Hcat + (size_t)(q0 + w * 16 + ln) * (H_ * HD_);

  for (int kh = 0; kh < 2; ++kh) {
    __syncthreads();
    { // stage W_O block [64 out][256 k] fp32 -> bf16
      const int row = tid >> 2, c0 = (tid & 3) * 64;
      const float4* s4 =
          (const float4*)(WO + (size_t)(cb + row) * FIN_ + kh * 256 + c0);
#pragma unroll
      for (int i = 0; i < 16; ++i) {
        float4 v = s4[i];
        bf16_t* d = &WOs[row][c0 + i * 4];
        d[0] = (bf16_t)v.x; d[1] = (bf16_t)v.y;
        d[2] = (bf16_t)v.z; d[3] = (bf16_t)v.w;
      }
    }
    __syncthreads();

#pragma unroll
    for (int kt = 0; kt < 8; ++kt) {
      const int kl = kt * 32;
      const int kg = kh * 256 + kl;
      v16bf a = frag_ld(arow + kg + hi8);
#pragma unroll
      for (int ct = 0; ct < 4; ++ct) {
        v16bf b = frag_ld(&WOs[ct * 16 + ln][kl + hi8]);
        C[ct] = wmma_bf16(a, b, C[ct]);
      }
    }
  }

#pragma unroll
  for (int ct = 0; ct < 4; ++ct)
#pragma unroll
    for (int r = 0; r < 8; ++r)
      out[(size_t)(q0 + w * 16 + r + hi8) * FOUT_ + cb + ct * 16 + ln] =
          C[ct][r];
}

// ---------- launch ----------
extern "C" void kernel_launch(void* const* d_in, const int* in_sizes, int n_in,
                              void* d_out, int out_size, void* d_ws, size_t ws_size,
                              hipStream_t stream) {
  const float* X    = (const float*)d_in[0];
  const int*   mask = (const int*)d_in[1];
  const float* WQ   = (const float*)d_in[2];
  const float* WK   = (const float*)d_in[3];
  const float* WV   = (const float*)d_in[4];
  const float* WO   = (const float*)d_in[5];
  float* out = (float*)d_out;

  unsigned char* wsb = (unsigned char*)d_ws;
  const size_t seg = (size_t)H_ * N_ * HD_ * sizeof(bf16_t); // 4 MB
  bf16_t* Qb   = (bf16_t*)(wsb);
  bf16_t* Kb   = (bf16_t*)(wsb + seg);
  bf16_t* VTb  = (bf16_t*)(wsb + 2 * seg);
  bf16_t* Hcat = (bf16_t*)(wsb + 3 * seg); // N x 512 bf16 = 4 MB  (total 16 MB)

  qkv_kernel<<<dim3(N_ / 128, H_), 256, 0, stream>>>(X, WQ, WK, WV, Qb, Kb, VTb);
  attn_kernel<<<dim3(N_ / 128, H_), 256, 0, stream>>>(mask, Qb, Kb, VTb, Hcat);
  proj_kernel<<<dim3(N_ / 128, FOUT_ / 64), 256, 0, stream>>>(WO, Hcat, out);
}
